// JunmaiLayer_68582037782789
// MI455X (gfx1250) — matrix operand order repaired and checked by reading, not compiled, not run on hardware
//
#include <hip/hip_runtime.h>
#include <hip/hip_bf16.h>
#include <math.h>

typedef __attribute__((ext_vector_type(16))) _Float16 v16h;
typedef __attribute__((ext_vector_type(8)))  float    v8f;

#define N_ATOMS 768
#define F_DIM   64
#define C_DIM   64
#define R_RBF   50
#define PI_OVER_CUT 0.6283185307179586f   /* pi / 5 */

// ---------------------------------------------------------------------------
// Kernel 1: decompose the concat(h_i,h_j)@W_node GEMM into four 768x64 blocks:
//   KA[i,c] = h[i]@W_node[:64,   c]      + b_node[c]
//   QA[i,c] = h[i]@W_node[:64,   64+c]   + b_node[64+c]
//   KB[j,c] = h[j]@W_node[64:,   c]
//   QB[j,c] = h[j]@W_node[64:,   64+c]
// so K[i,j,c] = KA[i,c]+KB[j,c],  Q[i,j,c] = QA[i,c]+QB[j,c].
// Tiny (25 MFLOP): plain fp32.
// ---------------------------------------------------------------------------
__global__ __launch_bounds__(256)
void kq_precompute(const float* __restrict__ h, const float* __restrict__ W_node,
                   const float* __restrict__ b_node,
                   float* __restrict__ KA, float* __restrict__ QA,
                   float* __restrict__ KB, float* __restrict__ QB) {
  const int i = blockIdx.x;
  const int tid = threadIdx.x;
  __shared__ float hs[F_DIM];
  if (tid < F_DIM) hs[tid] = h[i * F_DIM + tid];
  __syncthreads();

  const int which  = tid >> 6;          // 0:KA 1:QA 2:KB 3:QB
  const int c      = tid & 63;
  const int rowoff = (which >= 2) ? F_DIM : 0;
  const int col    = (which & 1) ? (C_DIM + c) : c;

  float acc = 0.f;
  #pragma unroll 8
  for (int f = 0; f < F_DIM; ++f)
    acc += hs[f] * W_node[(f + rowoff) * (2 * C_DIM) + col];

  if (which == 0) acc += b_node[c];
  if (which == 1) acc += b_node[C_DIM + c];

  float* dst = (which == 0) ? KA : (which == 1) ? QA : (which == 2) ? KB : QB;
  dst[i * C_DIM + c] = acc;
}

// ---------------------------------------------------------------------------
// Kernel 2: fully fused per-row pass. One block per i (8 waves).
// Each wave owns a 16-j tile per iteration:
//   - computes d, cutoff, exp(-d), sumsq per j
//   - builds smear(16x64, K-padded) A-fragments IN REGISTERS (ISA 7.12.2
//     16-bit A layout), W_basis B-fragments preloaded in LDS fragment layout
//   - 2 x v_wmma_f32_16x16x32_f16 per c-tile -> basis(16x16) in D layout
//   - fused elementwise K*Q*basis^2*sumsq, reduced over j into hn_s[64]
// Then the 64x64 tanh-MLP head, all without touching any N^2 tensor.
// ---------------------------------------------------------------------------
__global__ __launch_bounds__(256)
void fused_pair_kernel(const float* __restrict__ x,
                       const float* __restrict__ W_basis,
                       const float* __restrict__ means,
                       const float* __restrict__ betas,
                       const float* __restrict__ KA, const float* __restrict__ QA,
                       const float* __restrict__ KB, const float* __restrict__ QB,
                       const float* __restrict__ W_s1, const float* __restrict__ b_s1,
                       const float* __restrict__ W_s2, const float* __restrict__ b_s2,
                       float* __restrict__ out) {
  const int i    = blockIdx.x;
  const int tid  = threadIdx.x;
  const int lane = tid & 31;
  const int wave = tid >> 5;
  const bool hihalf = (lane >= 16);

  __shared__ __align__(32) _Float16 Bfrag[2][4][32][16]; // [kstep][ctile][lane][16 halves]
  __shared__ float hn_s[C_DIM];
  __shared__ float KAi_s[C_DIM], QAi_s[C_DIM];
  __shared__ float means_s[64], betas_s[64];
  __shared__ float sumsq_s[8][16];
  __shared__ float act_s[C_DIM];

  // --- Stage W_basis as WMMA B-fragments (B 32x16 f16 layout, ISA 7.12.2):
  // lane<16: col n=lane, K = {2v,2v+1} + (v>=4 ? 8 : 0) ; lane>=16: col n=lane-16, same K + 8
  {
    const int s  = tid >> 7;           // k-step (0..1)
    const int ct = (tid >> 5) & 3;     // c-tile (0..3)
    const int lf = tid & 31;
    const int n  = ct * 16 + (lf & 15);
    #pragma unroll
    for (int v = 0; v < 8; ++v) {
      const int k = 32 * s + 2 * v + ((v >= 4) ? 8 : 0) + ((lf >= 16) ? 8 : 0);
      Bfrag[s][ct][lf][2*v]   = (k     < R_RBF) ? (_Float16)W_basis[k * C_DIM + n]       : (_Float16)0.f;
      Bfrag[s][ct][lf][2*v+1] = (k + 1 < R_RBF) ? (_Float16)W_basis[(k + 1) * C_DIM + n] : (_Float16)0.f;
    }
  }
  if (tid < C_DIM) {
    KAi_s[tid]  = KA[i * C_DIM + tid];
    QAi_s[tid]  = QA[i * C_DIM + tid];
    hn_s[tid]   = 0.f;
    means_s[tid] = (tid < R_RBF) ? means[tid] : 0.f;
    betas_s[tid] = (tid < R_RBF) ? betas[tid] : 0.f;
  }
  __syncthreads();

  const float xi0 = x[i * 3 + 0];
  const float xi1 = x[i * 3 + 1];
  const float xi2 = x[i * 3 + 2];

  float hn_part[4] = {0.f, 0.f, 0.f, 0.f};

  // 768 j's / (8 waves * 16) = 6 uniform iterations -> EXEC all-ones at WMMA
  for (int iter = 0; iter < N_ATOMS / (16 * 8); ++iter) {
    const int jbase = (iter * 8 + wave) * 16;
    const int m = lane & 15;            // A-matrix row this lane feeds
    const int j = jbase + m;

    const float dx0 = xi0 - x[j * 3 + 0];
    const float dx1 = xi1 - x[j * 3 + 1];
    const float dx2 = xi2 - x[j * 3 + 2];
    const float d2  = dx0 * dx0 + dx1 * dx1 + dx2 * dx2;
    const float d   = sqrtf(d2 + 1e-10f);
    const float denom = d * d + 1e-10f;
    const float sumsq = d2 / (denom * denom);
    const float cutoff = (d < 5.0f) ? 0.5f * (cosf(d * PI_OVER_CUT) + 1.0f) : 0.f;
    const float ed = expf(-d);
    if (!hihalf) sumsq_s[wave][m] = sumsq;   // same-wave LDS in-order: later reads see this

    // --- A fragments in registers (16-bit A 16x32 layout):
    // VGPR v holds K = {2v,2v+1} + (v>=4 ? 8 : 0) + (lane>=16 ? 8 : 0)
    v16h a0, a1;
    #pragma unroll
    for (int v = 0; v < 8; ++v) {
      const int kb = 2 * v + ((v >= 4) ? 8 : 0) + (hihalf ? 8 : 0);
      #pragma unroll
      for (int p = 0; p < 2; ++p) {
        const int r0 = kb + p;          // k-step 0
        const int r1 = kb + p + 32;     // k-step 1
        float s0 = 0.f, s1 = 0.f;
        if (r0 < R_RBF) { float t = ed - means_s[r0]; s0 = cutoff * expf(-betas_s[r0] * t * t); }
        if (r1 < R_RBF) { float t = ed - means_s[r1]; s1 = cutoff * expf(-betas_s[r1] * t * t); }
        a0[2 * v + p] = (_Float16)s0;
        a1[2 * v + p] = (_Float16)s1;
      }
    }

    // --- basis = smear @ W_basis  (4 c-tiles x 2 k-steps of WMMA), then fuse
    #pragma unroll
    for (int ct = 0; ct < 4; ++ct) {
      const v16h b0 = *(const v16h*)(&Bfrag[0][ct][lane][0]);
      const v16h b1 = *(const v16h*)(&Bfrag[1][ct][lane][0]);
      v8f acc = {};
      acc = __builtin_amdgcn_wmma_f32_16x16x32_f16(false, a0, false, b0, (short)0, acc, false, false);
      acc = __builtin_amdgcn_wmma_f32_16x16x32_f16(false, a1, false, b1, (short)0, acc, false, false);

      const int n = lane & 15;
      const int c = ct * 16 + n;
      const float kai = KAi_s[c];
      const float qai = QAi_s[c];
      float part = 0.f;
      #pragma unroll
      for (int v = 0; v < 8; ++v) {
        const int m2 = v + (hihalf ? 8 : 0);  // D layout: row = v + 8*hi
        const int j2 = jbase + m2;
        const float basis = acc[v];
        const float kq = (kai + KB[j2 * C_DIM + c]) * (qai + QB[j2 * C_DIM + c]);
        part += kq * basis * basis * sumsq_s[wave][m2];
      }
      hn_part[ct] += part;
    }
  }

  #pragma unroll
  for (int ct = 0; ct < 4; ++ct)
    atomicAdd(&hn_s[ct * 16 + (lane & 15)], hn_part[ct]);
  __syncthreads();

  // --- head: out[i] = tanh(hn @ W_s1 + b1) @ W_s2 + b2
  if (tid < C_DIM) {
    float s = b_s1[tid];
    #pragma unroll 8
    for (int c = 0; c < C_DIM; ++c) s += hn_s[c] * W_s1[c * C_DIM + tid];
    act_s[tid] = tanhf(s);
  }
  __syncthreads();
  if (tid < C_DIM) {
    float o = b_s2[tid];
    #pragma unroll 8
    for (int c = 0; c < C_DIM; ++c) o += act_s[c] * W_s2[c * C_DIM + tid];
    out[i * C_DIM + tid] = o;
  }
}

// ---------------------------------------------------------------------------
extern "C" void kernel_launch(void* const* d_in, const int* in_sizes, int n_in,
                              void* d_out, int out_size, void* d_ws, size_t ws_size,
                              hipStream_t stream) {
  const float* h       = (const float*)d_in[0];
  const float* x       = (const float*)d_in[1];
  const float* W_node  = (const float*)d_in[2];
  const float* b_node  = (const float*)d_in[3];
  const float* W_basis = (const float*)d_in[4];
  const float* means   = (const float*)d_in[5];
  const float* betas   = (const float*)d_in[6];
  const float* W_s1    = (const float*)d_in[7];
  const float* b_s1    = (const float*)d_in[8];
  const float* W_s2    = (const float*)d_in[9];
  const float* b_s2    = (const float*)d_in[10];
  float* out = (float*)d_out;

  float* KA = (float*)d_ws;                 // 4 x 768 x 64 floats = 768 KB scratch
  float* QA = KA + N_ATOMS * C_DIM;
  float* KB = QA + N_ATOMS * C_DIM;
  float* QB = KB + N_ATOMS * C_DIM;

  kq_precompute<<<N_ATOMS, 256, 0, stream>>>(h, W_node, b_node, KA, QA, KB, QB);
  fused_pair_kernel<<<N_ATOMS, 256, 0, stream>>>(x, W_basis, means, betas,
                                                 KA, QA, KB, QB,
                                                 W_s1, b_s1, W_s2, b_s2, out);
}
